// ConvShiftLayer_52072183496865
// MI455X (gfx1250) — compile-verified, matrix operand order checked by reading, not run on hardware
//
#include <hip/hip_runtime.h>
#include <hip/hip_bf16.h>

// ConvShiftLayer for MI455X (gfx1250, wave32), 4 passes:
//   1) split x (f32) -> x_hi + x_lo (bf16 RNE pair, ~16 effective mantissa bits)
//   2) split + transpose W -> Wt_hi/Wt_lo [N,K] bf16 (GEMM B-fragments K-contiguous)
//   3) h = x@W + b via v_wmma_f32_16x16x32_bf16 as ONE GEMM with K' = 3K:
//      segments [x_hi*W_hi | x_hi*W_lo | x_lo*W_hi], f32 accumulate.
//      (dropped lo*lo term ~2^-16 relative; ~1e-4 after K=1024 accumulation)
//   4) y[t,f] = sum_{j=0..7} h[t-j,(f+j)%F]; a=tanh(y); z=(a>0)
// GEMM dominates (68.7 GFLOP). Stage slabs are DMA'd straight into LDS with
// gfx1250 async loads (global_load_async_to_lds_b128, ASYNCcnt) so the single
// s_wait_asynccnt 0 lands AFTER a full 8-WMMA compute phase — no loadcnt stall
// at the barrier, no VGPR staging, no ds_store commits in the wave DS stream.

typedef __attribute__((ext_vector_type(8)))  float  v8f;
typedef __attribute__((ext_vector_type(16))) __bf16 v16bf;

#define TM   128
#define TN   128
#define TKB  32    // K per LDS stage == bf16 WMMA K
#define LDSP 40    // LDS row stride in bf16 elems (80B: 16B-aligned, 5 coprime 16)

union Frag16 { v16bf v; uint4 q[2]; };

// ---------------- pass 1: elementwise hi/lo split of x ----------------
__global__ __launch_bounds__(256) void split_x_bf16(
    const float* __restrict__ src,
    __hip_bfloat16* __restrict__ hi, __hip_bfloat16* __restrict__ lo, int n)
{
    const int i = blockIdx.x * 256 + threadIdx.x;
    if (i >= n) return;
    const float v = src[i];
    const __hip_bfloat16 h = __float2bfloat16(v);
    hi[i] = h;
    lo[i] = __float2bfloat16(v - __bfloat162float(h));
}

// ---------------- pass 2: split + transpose W [K,N] -> Wt [N,K] ----------------
__global__ __launch_bounds__(256) void split_transpose_w_bf16(
    const float* __restrict__ W,
    __hip_bfloat16* __restrict__ WtHi, __hip_bfloat16* __restrict__ WtLo,
    int K, int N)
{
    __shared__ float tile[32][33];
    const int tx = threadIdx.x & 31;
    const int ty = threadIdx.x >> 5;           // 0..7
    const int n0 = blockIdx.x * 32;
    const int k0 = blockIdx.y * 32;
    #pragma unroll
    for (int r = 0; r < 32; r += 8)            // coalesced along n
        tile[ty + r][tx] = W[(size_t)(k0 + ty + r) * N + (n0 + tx)];
    __syncthreads();
    #pragma unroll
    for (int r = 0; r < 32; r += 8) {          // coalesced along k
        const float v = tile[tx][ty + r];      // = W[k0+tx][n0+ty+r]
        const __hip_bfloat16 h = __float2bfloat16(v);
        const size_t o = (size_t)(n0 + ty + r) * K + (k0 + tx);
        WtHi[o] = h;
        WtLo[o] = __float2bfloat16(v - __bfloat162float(h));
    }
}

// ---------------- pass 3: bf16x3 WMMA GEMM, h = x@W + b ----------------
__global__ __launch_bounds__(256) void gemm_bf16x3(
    const __hip_bfloat16* __restrict__ Xhi, const __hip_bfloat16* __restrict__ Xlo,
    const __hip_bfloat16* __restrict__ WtHi, const __hip_bfloat16* __restrict__ WtLo,
    const float* __restrict__ bias, float* __restrict__ H,
    int M, int N, int K)                       // K = per-segment depth (1024)
{
    __shared__ unsigned short As[2][TM][LDSP]; // 20 KB
    __shared__ unsigned short Bs[2][TN][LDSP]; // 20 KB

    const int tid  = threadIdx.x;
    const int wave = tid >> 5;
    const int lane = tid & 31;
    const int half = lane >> 4;                // K-chunk / output-row-half select
    const int lh   = lane & 15;

    const int m0 = blockIdx.x * TM;
    const int n0 = blockIdx.y * TN;
    const int wm = (wave & 3) * 32;            // 2 M-tiles of 16
    const int wn = (wave >> 2) * 64;           // 4 N-tiles of 16

    v8f acc[2][4] = {};

    // per-thread cooperative-copy coordinates (2 x 16B chunks per slab per thread)
    const int lrow = tid >> 2;                 // 0..63
    const int lcol = (tid & 3) * 8;            // bf16 elems: 0,8,16,24
    const size_t aOff0 = (size_t)(m0 + lrow)      * K + lcol;
    const size_t aOff1 = (size_t)(m0 + lrow + 64) * K + lcol;
    const size_t bOff0 = (size_t)(n0 + lrow)      * K + lcol;
    const size_t bOff1 = (size_t)(n0 + lrow + 64) * K + lcol;

    // per-thread LDS destination addresses (raw 32-bit LDS offsets) per buffer
    unsigned ldsA0[2], ldsA1[2], ldsB0[2], ldsB1[2];
    #pragma unroll
    for (int s = 0; s < 2; ++s) {
        ldsA0[s] = (unsigned)(uintptr_t)&As[s][lrow][lcol];
        ldsA1[s] = (unsigned)(uintptr_t)&As[s][lrow + 64][lcol];
        ldsB0[s] = (unsigned)(uintptr_t)&Bs[s][lrow][lcol];
        ldsB1[s] = (unsigned)(uintptr_t)&Bs[s][lrow + 64][lcol];
    }

    const int nStages = 3 * (K / TKB);         // 96 (K' = 3K concat trick), even

    // gfx1250 async DMA: global -> LDS, tracked by ASYNCcnt (not LOADcnt/DScnt)
    auto issue_async = [&](int stage, int s) {
        const int seg = stage >> 5;            // 0: hi*hi  1: hi*lo  2: lo*hi
        const int k0  = (stage & 31) << 5;
        const __hip_bfloat16* Aseg = (seg == 2) ? Xlo  : Xhi;
        const __hip_bfloat16* Bseg = (seg == 1) ? WtLo : WtHi;
        const __hip_bfloat16* ga0 = Aseg + aOff0 + k0;
        const __hip_bfloat16* ga1 = Aseg + aOff1 + k0;
        const __hip_bfloat16* gb0 = Bseg + bOff0 + k0;
        const __hip_bfloat16* gb1 = Bseg + bOff1 + k0;
        asm volatile(
            "global_load_async_to_lds_b128 %0, %4, off\n\t"
            "global_load_async_to_lds_b128 %1, %5, off\n\t"
            "global_load_async_to_lds_b128 %2, %6, off\n\t"
            "global_load_async_to_lds_b128 %3, %7, off"
            :
            : "v"(ldsA0[s]), "v"(ldsA1[s]), "v"(ldsB0[s]), "v"(ldsB1[s]),
              "v"(ga0), "v"(ga1), "v"(gb0), "v"(gb1)
            : "memory");
    };

    auto compute = [&](int s) {
        // ISA 16-bit fragment layout (A 16x32 / B 32x16, both K-contiguous/lane):
        //   lanes 0-15:  K[0..7]  in v0-3, K[16..23] in v4-7
        //   lanes 16-31: K[8..15] in v0-3, K[24..31] in v4-7
        const int kc = 8 * half;
        Frag16 af[2];
        #pragma unroll
        for (int mt = 0; mt < 2; ++mt) {
            const unsigned short* p = &As[s][wm + mt * 16 + lh][kc];
            af[mt].q[0] = *(const uint4*)p;
            af[mt].q[1] = *(const uint4*)(p + 16);
        }
        Frag16 bf[4];
        #pragma unroll
        for (int nt = 0; nt < 4; ++nt) {
            const unsigned short* p = &Bs[s][wn + nt * 16 + lh][kc];
            bf[nt].q[0] = *(const uint4*)p;
            bf[nt].q[1] = *(const uint4*)(p + 16);
        }
        #pragma unroll
        for (int mt = 0; mt < 2; ++mt)
            #pragma unroll
            for (int nt = 0; nt < 4; ++nt)
                acc[mt][nt] = __builtin_amdgcn_wmma_f32_16x16x32_bf16(
                    false, af[mt].v, false, bf[nt].v,
                    (short)0, acc[mt][nt], false, false);
    };

    // prologue: DMA stage 0 into buf 0 and publish it
    issue_async(0, 0);
    asm volatile("s_wait_asynccnt 0x0" ::: "memory");
    __syncthreads();

    // main loop, unrolled x2 so the LDS buffer index is compile-time.
    // Async writes to buf b are issued only after the barrier that closed the
    // compute phase reading buf b; reads happen after s_wait_asynccnt + barrier.
    for (int st = 0; st < nStages; st += 2) {
        if (st + 1 < nStages) issue_async(st + 1, 1);  // DMA while we compute
        compute(0);
        asm volatile("s_wait_asynccnt 0x0" ::: "memory");
        __syncthreads();

        if (st + 2 < nStages) issue_async(st + 2, 0);
        compute(1);
        asm volatile("s_wait_asynccnt 0x0" ::: "memory");
        __syncthreads();
    }

    // C/D layout: VGPR r holds row (r + 8*half), col = lh (+tile offsets)
    #pragma unroll
    for (int mt = 0; mt < 2; ++mt) {
        #pragma unroll
        for (int nt = 0; nt < 4; ++nt) {
            const int col = n0 + wn + nt * 16 + lh;
            const float bv = bias[col];
            #pragma unroll
            for (int r = 0; r < 8; ++r) {
                const int row = m0 + wm + mt * 16 + half * 8 + r;
                H[(size_t)row * N + col] = acc[mt][nt][r] + bv;
            }
        }
    }
}

// ---------------- pass 4: diagonal shift-conv + tanh + sign ----------------
__global__ __launch_bounds__(256) void convshift_epilogue(
    const float* __restrict__ H,  // [M,F], M = B*L
    float* __restrict__ Y, float* __restrict__ A, float* __restrict__ Z,
    int L, int F, int total)
{
    const int idx = blockIdx.x * 256 + threadIdx.x;
    if (idx >= total) return;
    const int f   = idx % F;
    const int row = idx / F;
    const int t   = row % L;                   // taps never cross batches

    float y = 0.0f;
    #pragma unroll
    for (int j = 0; j < 8; ++j) {
        if (t - j < 0) break;                  // zero-padded region
        int ff = f + j;
        if (ff >= F) ff -= F;                  // jnp.roll wrap on feature axis
        y += H[(size_t)(row - j) * F + ff];
    }
    const float a = tanhf(y);
    Y[idx] = y;
    A[idx] = a;
    Z[idx] = (a > 0.0f) ? 1.0f : 0.0f;
}

extern "C" void kernel_launch(void* const* d_in, const int* in_sizes, int n_in,
                              void* d_out, int out_size, void* d_ws, size_t ws_size,
                              hipStream_t stream) {
    const float* x    = (const float*)d_in[0];   // (8,2048,1024) f32
    const float* Wd   = (const float*)d_in[1];   // (1024,2048)   f32
    const float* bias = (const float*)d_in[2];   // (2048,)       f32

    const int F = in_sizes[2];               // 2048
    const int K = in_sizes[1] / F;           // 1024
    const int M = in_sizes[0] / K;           // 16384
    const int L = 2048;                      // per reference
    const int N = F;

    // workspace layout: H (f32) | Xhi | Xlo | WtHi | WtLo  (~206 MB total)
    char* ws = (char*)d_ws;
    float* H = (float*)ws;
    size_t off = (size_t)M * N * sizeof(float);
    __hip_bfloat16* Xhi  = (__hip_bfloat16*)(ws + off);
    __hip_bfloat16* Xlo  = Xhi + (size_t)M * K;
    __hip_bfloat16* WtHi = Xlo + (size_t)M * K;
    __hip_bfloat16* WtLo = WtHi + (size_t)N * K;

    const int nx = M * K;                    // 16,777,216
    split_x_bf16<<<(nx + 255) / 256, 256, 0, stream>>>(x, Xhi, Xlo, nx);

    dim3 gT(N / 32, K / 32);                 // (64, 32)
    split_transpose_w_bf16<<<gT, 256, 0, stream>>>(Wd, WtHi, WtLo, K, N);

    dim3 g1(M / TM, N / TN);                 // (128, 16)
    gemm_bf16x3<<<g1, 256, 0, stream>>>(Xhi, Xlo, WtHi, WtLo, bias, H, M, N, K);

    const int total = M * N;                 // 33,554,432 per output tensor
    float* out = (float*)d_out;
    convshift_epilogue<<<(total + 255) / 256, 256, 0, stream>>>(
        H, out, out + (size_t)total, out + 2 * (size_t)total, L, F, total);
}
